// GAT0tapfc1_33200097198306
// MI455X (gfx1250) — matrix operand order repaired and checked by reading, not compiled, not run on hardware
//
#include <hip/hip_runtime.h>
#include <hip/hip_bf16.h>

#define HD __device__ __forceinline__

typedef __attribute__((ext_vector_type(16))) __bf16 v16bf;
typedef __attribute__((ext_vector_type(8)))  __bf16 v8bf;
typedef __attribute__((ext_vector_type(8)))  float  v8f;
typedef __attribute__((ext_vector_type(4)))  float  v4f;

// Problem sizes (match reference)
static constexpr int N_NODES = 16384;
static constexpr int N_EDGES = 131072;
static constexpr int ED      = 16;        // edge feature dim
static constexpr int HEADS   = 4;
static constexpr int O1 = 256, DD1 = HEADS * O1;   // 1024
static constexpr int O2 = 64,  DD2 = HEADS * O2;   // 256
static constexpr int NGRAPH  = 32;
static constexpr int E2      = N_EDGES + N_NODES;  // edges + self loops

// ---------------------------------------------------------------- utilities
__global__ void k_fill(float* __restrict__ p, long long n, float v) {
    long long i  = (long long)blockIdx.x * blockDim.x + threadIdx.x;
    long long st = (long long)gridDim.x * blockDim.x;
    for (; i < n; i += st) p[i] = v;
}

HD void atomicMaxFloat(float* addr, float val) {
    int* ai = (int*)addr;
    int old = *ai;
    while (__int_as_float(old) < val) {
        int prev = atomicCAS(ai, old, __float_as_int(val));
        if (prev == old) break;
        old = prev;
    }
}

// fp32 -> bf16, 8 elements/thread, vectorized (b128 in, b128 out)
__global__ void k_cvt_bf16(const float* __restrict__ in, __bf16* __restrict__ out,
                           long long n8) {               // n8 = n/8
    long long i  = (long long)blockIdx.x * blockDim.x + threadIdx.x;
    long long st = (long long)gridDim.x * blockDim.x;
    for (; i < n8; i += st) {
        v4f a = ((const v4f*)in)[2 * i];
        v4f b = ((const v4f*)in)[2 * i + 1];
        v8bf o;
#pragma unroll
        for (int k = 0; k < 4; ++k) { o[k] = (__bf16)a[k]; o[4 + k] = (__bf16)b[k]; }
        ((v8bf*)out)[i] = o;
    }
}

// W[K,Nc] fp32 -> Wt[Nc,K] bf16 (transposed so B-fragments are contiguous rows)
__global__ void k_cvt_w_t(const float* __restrict__ W, __bf16* __restrict__ Wt,
                          int K, int Nc) {
    int i  = blockIdx.x * blockDim.x + threadIdx.x;
    int st = gridDim.x * blockDim.x;
    int total = K * Nc;
    for (; i < total; i += st) {
        int k = i / Nc, c = i % Nc;
        Wt[(size_t)c * K + k] = (__bf16)W[i];
    }
}

// ------------------------------------------------- self-loop attr (mean of incoming)
__global__ void k_degree(const int* __restrict__ dst, const float* __restrict__ eattr,
                         float* __restrict__ deg, float* __restrict__ loopa, int E) {
    int e = blockIdx.x * blockDim.x + threadIdx.x;
    if (e >= E) return;
    int d = dst[e];
    atomicAdd(&deg[d], 1.0f);
    const float* ea = eattr + (size_t)e * ED;
    float* lp = loopa + (size_t)d * ED;
#pragma unroll
    for (int k = 0; k < ED; ++k) atomicAdd(&lp[k], ea[k]);
}

__global__ void k_loopdiv(float* __restrict__ loopa, const float* __restrict__ deg, int n) {
    int i = blockIdx.x * blockDim.x + threadIdx.x;
    if (i >= n * ED) return;
    loopa[i] /= fmaxf(deg[i / ED], 1.0f);
}

// ---------------------------------------------------------------- WMMA GEMM
// C[M,Nc] = A[M,K] @ W[K,Nc] + bias.  A: bf16 row-major. Wt: bf16 [Nc,K] (W^T).
// One wave -> 16(M) x 64(N) C tile; 2xb128 loads per fragment; fp32 accumulate.
HD v16bf join16(v8bf lo, v8bf hi) {
    v16bf r;
#pragma unroll
    for (int i = 0; i < 8; ++i) { r[i] = lo[i]; r[8 + i] = hi[i]; }
    return r;
}

HD v16bf load_a_frag(const __bf16* __restrict__ A, int K, int mBase, int k0, int lane) {
    int row  = mBase + (lane & 15);
    int koff = (lane < 16) ? 0 : 8;          // ISA 16-bit A layout (16x32)
    const __bf16* ap = A + (size_t)row * K + k0 + koff;
    return join16(*(const v8bf*)ap, *(const v8bf*)(ap + 16));
}

HD v16bf load_b_frag(const __bf16* __restrict__ Wt, int K, int k0, int nBase, int lane) {
    int col = nBase + (lane & 15);
    int kb  = (lane < 16) ? 0 : 8;           // ISA 16-bit B layout (32x16)
    const __bf16* bp = Wt + (size_t)col * K + k0 + kb;
    return join16(*(const v8bf*)bp, *(const v8bf*)(bp + 16));
}

__global__ void k_gemm_bias_wmma(const __bf16* __restrict__ A, const __bf16* __restrict__ Wt,
                                 const float* __restrict__ bias, float* __restrict__ C,
                                 int M, int K, int Nc) {
    int lane   = threadIdx.x & 31;
    int wave   = (blockIdx.x * blockDim.x + threadIdx.x) >> 5;
    int mTiles = M >> 4;
    int nGrp   = Nc >> 6;
    if (wave >= mTiles * nGrp) return;       // wave-uniform: EXEC all-1 for WMMA
    int mBase = (wave % mTiles) << 4;        // consecutive waves share the N-group
    int nBase = (wave / mTiles) << 6;        //  -> B fragments hit L0/L2

    v8f c0 = {}, c1 = {}, c2 = {}, c3 = {};
    for (int k0 = 0; k0 < K; k0 += 32) {
        v16bf a  = load_a_frag(A, K, mBase, k0, lane);
        v16bf b0 = load_b_frag(Wt, K, k0, nBase,      lane);
        v16bf b1 = load_b_frag(Wt, K, k0, nBase + 16, lane);
        v16bf b2 = load_b_frag(Wt, K, k0, nBase + 32, lane);
        v16bf b3 = load_b_frag(Wt, K, k0, nBase + 48, lane);
        c0 = __builtin_amdgcn_wmma_f32_16x16x32_bf16(false, a, false, b0, (short)0, c0, false, false);
        c1 = __builtin_amdgcn_wmma_f32_16x16x32_bf16(false, a, false, b1, (short)0, c1, false, false);
        c2 = __builtin_amdgcn_wmma_f32_16x16x32_bf16(false, a, false, b2, (short)0, c2, false, false);
        c3 = __builtin_amdgcn_wmma_f32_16x16x32_bf16(false, a, false, b3, (short)0, c3, false, false);
    }
    // 32-bit C/D layout: VGPR r -> M=r (lanes 0-15) / M=8+r (lanes 16-31), N=lane%16
    int rbase = (lane < 16) ? 0 : 8;
    int col   = nBase + (lane & 15);
#pragma unroll
    for (int r = 0; r < 8; ++r) {
        size_t row = (size_t)(mBase + rbase + r) * Nc;
        C[row + col]      = c0[r] + bias[col];
        C[row + col + 16] = c1[r] + bias[col + 16];
        C[row + col + 32] = c2[r] + bias[col + 32];
        C[row + col + 48] = c3[r] + bias[col + 48];
    }
}

// ------------------------------------------------- edge attention, 3 passes
// Pass 1: logits[e,h] = sum_d lrelu(xl[src]+xr[dst]+(eattr@We)) . att ; segment max
__global__ void k_edge_logits(const int* __restrict__ src, const int* __restrict__ dst,
                              const float* __restrict__ eattr, const float* __restrict__ loopa,
                              const float* __restrict__ xl, const float* __restrict__ xr,
                              const float* __restrict__ We, const float* __restrict__ att,
                              float* __restrict__ lg, float* __restrict__ mx,
                              int E, int Nn, int D, int O) {
    extern __shared__ float sWe[];                      // [ED, D] <= 64KB
    for (int i = threadIdx.x; i < ED * D; i += blockDim.x) sWe[i] = We[i];
    __syncthreads();

    long long total = (long long)(E + Nn) * HEADS;
    long long st = (long long)gridDim.x * blockDim.x;
    for (long long idx = (long long)blockIdx.x * blockDim.x + threadIdx.x; idx < total; idx += st) {
        int e = (int)(idx >> 2);
        int h = (int)(idx & 3);
        int s, d; const float* ea;
        if (e < E) { s = src[e]; d = dst[e]; ea = eattr + (size_t)e * ED; }
        else       { s = d = e - E;          ea = loopa + (size_t)(e - E) * ED; }
        float eav[ED];
#pragma unroll
        for (int k = 0; k < ED; ++k) eav[k] = ea[k];

        const float* xls = xl + (size_t)s * D + h * O;
        const float* xrd = xr + (size_t)d * D + h * O;
        const float* weh = sWe + h * O;
        const float* ah  = att + h * O;
        float acc = 0.f;
        for (int c = 0; c < O; ++c) {
            float ef = 0.f;
#pragma unroll
            for (int k = 0; k < ED; ++k) ef = fmaf(eav[k], weh[k * D + c], ef);
            float m  = xls[c] + xrd[c] + ef;
            float sv = (m > 0.f) ? m : 0.2f * m;        // leaky_relu(0.2)
            acc = fmaf(sv, ah[c], acc);
        }
        lg[idx] = acc;
        atomicMaxFloat(&mx[d * HEADS + h], acc);
    }
}

// Pass 2: ex = exp(logit - mx[dst]); segment sum
__global__ void k_edge_expsum(const int* __restrict__ dst, float* __restrict__ lg,
                              const float* __restrict__ mx, float* __restrict__ den,
                              int E, int Nn) {
    long long total = (long long)(E + Nn) * HEADS;
    long long st = (long long)gridDim.x * blockDim.x;
    for (long long idx = (long long)blockIdx.x * blockDim.x + threadIdx.x; idx < total; idx += st) {
        int e = (int)(idx >> 2);
        int h = (int)(idx & 3);
        int d = (e < E) ? dst[e] : (e - E);
        float ex = __expf(lg[idx] - mx[d * HEADS + h]);
        lg[idx] = ex;
        atomicAdd(&den[d * HEADS + h], ex);
    }
}

// Pass 3: agg[dst] += (ex/den[dst]) * xl[src]   (dest is L2-resident; fp32 atomics)
__global__ void k_edge_scatter(const int* __restrict__ src, const int* __restrict__ dst,
                               const float* __restrict__ lg, const float* __restrict__ den,
                               const float* __restrict__ xl, float* __restrict__ agg,
                               int E, int Nn, int D, int O) {
    long long total = (long long)(E + Nn) * D;
    long long st = (long long)gridDim.x * blockDim.x;
    for (long long idx = (long long)blockIdx.x * blockDim.x + threadIdx.x; idx < total; idx += st) {
        int e = (int)(idx / D);
        int c = (int)(idx % D);
        int h = c / O;
        int s, d;
        if (e < E) { s = src[e]; d = dst[e]; } else { s = d = e - E; }
        float alpha = lg[(size_t)e * HEADS + h] / den[d * HEADS + h];
        atomicAdd(&agg[(size_t)d * D + c], alpha * xl[(size_t)s * D + c]);
    }
}

__global__ void k_bias_relu(float* __restrict__ h, const float* __restrict__ bias,
                            long long n, int D) {
    long long st = (long long)gridDim.x * blockDim.x;
    for (long long i = (long long)blockIdx.x * blockDim.x + threadIdx.x; i < n; i += st)
        h[i] = fmaxf(h[i] + bias[(int)(i % D)], 0.f);
}

// ------------------------------------------------- pooling + FC
__global__ void k_pool(const float* __restrict__ h, const int* __restrict__ batch,
                       float* __restrict__ g, int n, int D) {
    long long total = (long long)n * D;
    long long st = (long long)gridDim.x * blockDim.x;
    for (long long i = (long long)blockIdx.x * blockDim.x + threadIdx.x; i < total; i += st)
        atomicAdd(&g[(size_t)batch[(int)(i / D)] * D + (int)(i % D)], h[i]);
}

__global__ void k_fc(const float* __restrict__ g, const float* __restrict__ fcW,
                     const float* __restrict__ fcb, float* __restrict__ out) {
    int wave = (blockIdx.x * blockDim.x + threadIdx.x) >> 5;   // one wave32 per graph
    int lane = threadIdx.x & 31;
    if (wave >= NGRAPH) return;
    float acc = 0.f;
    for (int c = lane; c < DD2; c += 32) acc = fmaf(g[wave * DD2 + c], fcW[c], acc);
#pragma unroll
    for (int off = 16; off; off >>= 1) acc += __shfl_xor(acc, off, 32);
    if (lane == 0) out[wave] = fmaxf(acc + fcb[0], 0.f);
}

// ---------------------------------------------------------------- launcher
extern "C" void kernel_launch(void* const* d_in, const int* in_sizes, int n_in,
                              void* d_out, int out_size, void* d_ws, size_t ws_size,
                              hipStream_t stream) {
    const float* x     = (const float*)d_in[0];
    const int*   eidx  = (const int*)d_in[1];
    const float* eattr = (const float*)d_in[2];
    const int*   batch = (const int*)d_in[3];
    const float* Wl1   = (const float*)d_in[4];
    const float* bl1   = (const float*)d_in[5];
    const float* Wr1   = (const float*)d_in[6];
    const float* br1   = (const float*)d_in[7];
    const float* We1   = (const float*)d_in[8];
    const float* att1  = (const float*)d_in[9];
    const float* bias1 = (const float*)d_in[10];
    const float* Wl2   = (const float*)d_in[11];
    const float* bl2   = (const float*)d_in[12];
    const float* Wr2   = (const float*)d_in[13];
    const float* br2   = (const float*)d_in[14];
    const float* We2   = (const float*)d_in[15];
    const float* att2  = (const float*)d_in[16];
    const float* bias2 = (const float*)d_in[17];
    const float* fcW   = (const float*)d_in[18];
    const float* fcb   = (const float*)d_in[19];
    const int* src = eidx;
    const int* dst = eidx + N_EDGES;
    float* out = (float*)d_out;
    (void)in_sizes; (void)n_in; (void)out_size; (void)ws_size;

    // workspace carve-out: fp32 region then bf16 region (~290 MB total)
    float* ws = (float*)d_ws;
    size_t off = 0;
    auto carve = [&](size_t n) { float* p = ws + off; off += n; return p; };
    float* deg   = carve(N_NODES);
    float* loopa = carve((size_t)N_NODES * ED);
    float* xl1   = carve((size_t)N_NODES * DD1);
    float* xr1   = carve((size_t)N_NODES * DD1);
    float* agg1  = carve((size_t)N_NODES * DD1);   // becomes h1 after bias+relu
    float* xl2   = carve((size_t)N_NODES * DD2);
    float* xr2   = carve((size_t)N_NODES * DD2);
    float* agg2  = carve((size_t)N_NODES * DD2);   // becomes h2
    float* lg    = carve((size_t)E2 * HEADS);      // logits then exp
    float* mx    = carve((size_t)N_NODES * HEADS);
    float* den   = carve((size_t)N_NODES * HEADS);
    float* gpool = carve((size_t)NGRAPH * DD2);

    __bf16* bfbase = (__bf16*)(ws + off);
    size_t boff = 0;
    auto carveb = [&](size_t n) { __bf16* p = bfbase + boff; boff += n; return p; };
    __bf16* xbf   = carveb((size_t)N_NODES * 64);    // x in bf16
    __bf16* h1bf  = carveb((size_t)N_NODES * DD1);   // h1 in bf16
    __bf16* Wl1t  = carveb((size_t)64 * DD1);        // W^T bf16
    __bf16* Wr1t  = carveb((size_t)64 * DD1);
    __bf16* Wl2t  = carveb((size_t)DD1 * DD2);
    __bf16* Wr2t  = carveb((size_t)DD1 * DD2);

    const int T = 256;
    const float NEG = -3.402823466e38f;

    // self-loop attrs: mean of incoming edge_attr
    k_fill<<<1024, T, 0, stream>>>(deg, N_NODES, 0.f);
    k_fill<<<1024, T, 0, stream>>>(loopa, (long long)N_NODES * ED, 0.f);
    k_degree<<<(N_EDGES + T - 1) / T, T, 0, stream>>>(dst, eattr, deg, loopa, N_EDGES);
    k_loopdiv<<<(N_NODES * ED + T - 1) / T, T, 0, stream>>>(loopa, deg, N_NODES);

    // ---- bf16 staging: activations + transposed weights
    k_cvt_bf16<<<1024, T, 0, stream>>>(x, xbf, (long long)N_NODES * 64 / 8);
    k_cvt_w_t<<<256, T, 0, stream>>>(Wl1, Wl1t, 64, DD1);
    k_cvt_w_t<<<256, T, 0, stream>>>(Wr1, Wr1t, 64, DD1);
    k_cvt_w_t<<<1024, T, 0, stream>>>(Wl2, Wl2t, DD1, DD2);
    k_cvt_w_t<<<1024, T, 0, stream>>>(Wr2, Wr2t, DD1, DD2);

    // ---- layer 1 projections (bf16 WMMA): K=64, Nc=1024
    {
        int waves = (N_NODES / 16) * (DD1 / 64);
        int blocks = (waves + 7) / 8;
        k_gemm_bias_wmma<<<blocks, T, 0, stream>>>(xbf, Wl1t, bl1, xl1, N_NODES, 64, DD1);
        k_gemm_bias_wmma<<<blocks, T, 0, stream>>>(xbf, Wr1t, br1, xr1, N_NODES, 64, DD1);
    }
    // ---- layer 1 attention + aggregation
    k_fill<<<256, T, 0, stream>>>(mx, (long long)N_NODES * HEADS, NEG);
    k_fill<<<256, T, 0, stream>>>(den, (long long)N_NODES * HEADS, 0.f);
    k_fill<<<4096, T, 0, stream>>>(agg1, (long long)N_NODES * DD1, 0.f);
    k_edge_logits<<<2048, T, ED * DD1 * sizeof(float), stream>>>(
        src, dst, eattr, loopa, xl1, xr1, We1, att1, lg, mx, N_EDGES, N_NODES, DD1, O1);
    k_edge_expsum<<<2048, T, 0, stream>>>(dst, lg, mx, den, N_EDGES, N_NODES);
    k_edge_scatter<<<8192, T, 0, stream>>>(src, dst, lg, den, xl1, agg1,
                                           N_EDGES, N_NODES, DD1, O1);
    k_bias_relu<<<4096, T, 0, stream>>>(agg1, bias1, (long long)N_NODES * DD1, DD1);

    // ---- layer 2 projections (bf16 WMMA): K=1024, Nc=256
    k_cvt_bf16<<<2048, T, 0, stream>>>(agg1, h1bf, (long long)N_NODES * DD1 / 8);
    {
        int waves = (N_NODES / 16) * (DD2 / 64);
        int blocks = (waves + 7) / 8;
        k_gemm_bias_wmma<<<blocks, T, 0, stream>>>(h1bf, Wl2t, bl2, xl2, N_NODES, DD1, DD2);
        k_gemm_bias_wmma<<<blocks, T, 0, stream>>>(h1bf, Wr2t, br2, xr2, N_NODES, DD1, DD2);
    }
    // ---- layer 2 attention + aggregation
    k_fill<<<256, T, 0, stream>>>(mx, (long long)N_NODES * HEADS, NEG);
    k_fill<<<256, T, 0, stream>>>(den, (long long)N_NODES * HEADS, 0.f);
    k_fill<<<2048, T, 0, stream>>>(agg2, (long long)N_NODES * DD2, 0.f);
    k_edge_logits<<<2048, T, ED * DD2 * sizeof(float), stream>>>(
        src, dst, eattr, loopa, xl2, xr2, We2, att2, lg, mx, N_EDGES, N_NODES, DD2, O2);
    k_edge_expsum<<<2048, T, 0, stream>>>(dst, lg, mx, den, N_EDGES, N_NODES);
    k_edge_scatter<<<8192, T, 0, stream>>>(src, dst, lg, den, xl2, agg2,
                                           N_EDGES, N_NODES, DD2, O2);
    k_bias_relu<<<2048, T, 0, stream>>>(agg2, bias2, (long long)N_NODES * DD2, DD2);

    // ---- global add pool + FC + relu
    k_fill<<<32, T, 0, stream>>>(gpool, (long long)NGRAPH * DD2, 0.f);
    k_pool<<<2048, T, 0, stream>>>(agg2, batch, gpool, N_NODES, DD2);
    k_fc<<<4, T, 0, stream>>>(gpool, fcW, fcb, out);
}
